// hgt_rn_42090679501531
// MI455X (gfx1250) — matrix operand order repaired and checked by reading, not compiled, not run on hardware
//
#include <hip/hip_runtime.h>

// ---------------- constants ----------------
#define BB 64
#define LL 512
#define MM 320
#define NHH 8
#define DHH 40
#define DHP 64          // head dim padded to WMMA K granularity
#define FFF 2048
#define NLL 2000
#define NNODE 10000
#define DN_ 128
#define DD_ 32
#define DT_ 32
#define DL_ 64
#define DG_ 96
#define KCAT 352
#define KC  256         // K-chunk staged in LDS
#define KCP (KC + 8)    // padded row stride (bank-conflict-free b128 reads)

typedef __bf16 bf16;
typedef __attribute__((ext_vector_type(16))) __bf16 v16bf;
typedef __attribute__((ext_vector_type(8)))  __bf16 v8bf;
typedef __attribute__((ext_vector_type(8)))  float  v8f;

// ---------------- small helpers ----------------
__device__ inline float wred_sum(float v) {
    #pragma unroll
    for (int o = 16; o > 0; o >>= 1) v += __shfl_xor(v, o, 32);
    return v;
}
__device__ inline float wred_max(float v) {
    #pragma unroll
    for (int o = 16; o > 0; o >>= 1) v = fmaxf(v, __shfl_xor(v, o, 32));
    return v;
}
__device__ inline v16bf cat16(v8bf lo, v8bf hi8) {
    return __builtin_shufflevector(lo, hi8, 0, 1, 2, 3, 4, 5, 6, 7,
                                            8, 9, 10, 11, 12, 13, 14, 15);
}
// CDNA5 async global->LDS copy (16B per lane), tracked by ASYNCcnt.
__device__ inline void async_cp16(unsigned ldsOff, const bf16* src) {
    asm volatile("global_load_async_to_lds_b128 %0, %1, off"
                 :: "v"(ldsOff), "v"(src) : "memory");
}
__device__ inline void wait_async0() {
    asm volatile("s_wait_asynccnt 0x0" ::: "memory");
}

// ---------------- WMMA GEMM: C = act(alpha * A·W^T + bias + colbias) ----------------
// Requirements: K % 32 == 0, Mr % 32 == 0, all A rows valid, W buffer has
// >= roundup(Nc,32) rows (pad rows may hold garbage; their columns are never
// stored). Block: 128 threads = 4 waves, block tile 128x32. Per wave: 32x32 C
// tile (4 accumulators). The 32-row W panel is staged per K-chunk into LDS via
// async global->LDS copies and shared by all 4 waves (ds_load_b128 reads).
// Batched over gridDim.z, z decomposed as iz1 = z/zdiv, iz2 = z%zdiv.
__global__ void gemm_tn(const bf16* __restrict__ A, long lda, long a1, long a2,
                        const bf16* __restrict__ W, long ldw, long w1, long w2,
                        const float* __restrict__ bias,
                        const float* __restrict__ colb, long cb1, long cb2,
                        float alpha, int relu,
                        float* __restrict__ oF, bf16* __restrict__ oB,
                        long ldc, long cs1, long cs2,
                        int Mr, int Nc, int K, int zdiv)
{
    __shared__ bf16 sW[32 * KCP];
    int z = blockIdx.z;
    long iz1 = z / zdiv, iz2 = z % zdiv;
    int wave = threadIdx.x >> 5, lane = threadIdx.x & 31;
    int m0 = blockIdx.y * 128 + wave * 32;
    int n0 = blockIdx.x * 32;
    int hi = lane >> 4, l16 = lane & 15;
    bool live = m0 < Mr;                         // wave-uniform
    const bf16* Wpanel = W + iz1 * w1 + iz2 * w2 + (long)n0 * ldw;
    const bf16* arow0 = A + iz1 * a1 + iz2 * a2 + (long)(m0 + l16) * lda;
    const bf16* arow1 = arow0 + 16 * lda;
    const bf16* sB0 = sW + (long)l16 * KCP + hi * 16;
    const bf16* sB1 = sB0 + 16 * KCP;
    unsigned sWbase = (unsigned)(size_t)&sW[0];
    v8f acc00 = {}, acc01 = {}, acc10 = {}, acc11 = {};
    for (int k0 = 0; k0 < K; k0 += KC) {
        int kc = (K - k0 < KC) ? (K - k0) : KC;
        // ---- stage W panel chunk: 32 rows x kc, via async copies ----
        int cpr = kc >> 3;                       // 16B chunks per row
        int chunks = 32 * cpr;
        for (int c = threadIdx.x; c < chunks; c += 128) {
            int row = c / cpr;
            int ko  = (c - row * cpr) << 3;
            async_cp16(sWbase + (unsigned)(row * KCP + ko) * 2,
                       Wpanel + (long)row * ldw + k0 + ko);
        }
        wait_async0();
        __syncthreads();
        // ---- compute on staged chunk ----
        if (live) {
            #pragma unroll 2
            for (int kk = 0; kk < kc; kk += 32) {
                __builtin_prefetch(arow0 + k0 + kk + 64, 0, 1);
                __builtin_prefetch(arow1 + k0 + kk + 64, 0, 1);
                v16bf a0 = cat16(*(const v8bf*)(arow0 + k0 + kk + hi * 16),
                                 *(const v8bf*)(arow0 + k0 + kk + hi * 16 + 8));
                v16bf a1 = cat16(*(const v8bf*)(arow1 + k0 + kk + hi * 16),
                                 *(const v8bf*)(arow1 + k0 + kk + hi * 16 + 8));
                v16bf b0 = cat16(*(const v8bf*)(sB0 + kk),
                                 *(const v8bf*)(sB0 + kk + 8));
                v16bf b1 = cat16(*(const v8bf*)(sB1 + kk),
                                 *(const v8bf*)(sB1 + kk + 8));
                acc00 = __builtin_amdgcn_wmma_f32_16x16x32_bf16(false, a0, false, b0,
                                                                (short)0, acc00, false, false);
                acc01 = __builtin_amdgcn_wmma_f32_16x16x32_bf16(false, a0, false, b1,
                                                                (short)0, acc01, false, false);
                acc10 = __builtin_amdgcn_wmma_f32_16x16x32_bf16(false, a1, false, b0,
                                                                (short)0, acc10, false, false);
                acc11 = __builtin_amdgcn_wmma_f32_16x16x32_bf16(false, a1, false, b1,
                                                                (short)0, acc11, false, false);
            }
        }
        __syncthreads();                         // chunk reuse barrier
    }
    if (!live) return;
    long off_c = iz1 * cs1 + iz2 * cs2;
    #pragma unroll
    for (int mt = 0; mt < 2; ++mt) {
        #pragma unroll
        for (int nt = 0; nt < 2; ++nt) {
            int cn = n0 + nt * 16 + l16;
            if (cn < Nc) {
                float bval = bias ? bias[cn] : 0.0f;
                float cbv  = colb ? colb[iz1 * cb1 + iz2 * cb2 + cn] : 0.0f;
                v8f acc = mt ? (nt ? acc11 : acc10) : (nt ? acc01 : acc00);
                #pragma unroll
                for (int r = 0; r < 8; ++r) {
                    int cm = m0 + mt * 16 + r + hi * 8;
                    float v = acc[r] * alpha + bval + cbv;
                    if (relu) v = fmaxf(v, 0.0f);
                    long idx = off_c + (long)cm * ldc + cn;
                    if (oF) oF[idx] = v;
                    if (oB) oB[idx] = (bf16)v;
                }
            }
        }
    }
}

// ---------------- split QKV into padded per-head Q, K and transposed V ----------------
// qp/kp: [B*NH, L, DHP] (cols 40..63 zero); vpT: [B*NH, DHP, L] (rows 40..63 zero)
__global__ void k_split_qkv(const bf16* __restrict__ qkv,
                            bf16* __restrict__ qp, bf16* __restrict__ kp,
                            bf16* __restrict__ vpT)
{
    long i = (long)blockIdx.x * 256 + threadIdx.x;      // over B*NH*L*DHP
    if (i >= (long)BB * NHH * LL * DHP) return;
    int d = (int)(i & (DHP - 1));
    long rem = i >> 6;
    int l = (int)(rem % LL);
    long zz = rem / LL;                                  // b*NH + h
    int b = (int)(zz / NHH), h = (int)(zz % NHH);
    bf16 q, k, v;
    if (d < DHH) {
        const bf16* base = qkv + ((long)b * LL + l) * 960 + h * DHH + d;
        q = base[0]; k = base[MM]; v = base[2 * MM];
    } else {
        q = (bf16)0.0f; k = (bf16)0.0f; v = (bf16)0.0f;
    }
    qp[(zz * LL + l) * DHP + d] = q;
    kp[(zz * LL + l) * DHP + d] = k;
    vpT[(zz * DHP + d) * LL + l] = v;
}

// ---------------- fp32 -> bf16 conversion ----------------
__global__ void k_cvt(const float* in, bf16* out, long n) {
    long i = (long)blockIdx.x * 256 + threadIdx.x;
    if (i < n) out[i] = (bf16)in[i];
}

// ---------------- GAT: h = emb[local_nodes] @ gat_w ; e1 = h@a1 ; e2 = h@a2 ----------------
__global__ void k_gat_prep(const int* local_nodes, const float* node_emb, const float* gat_w,
                           const float* a1, const float* a2, float* h, float* e1, float* e2)
{
    int i = blockIdx.x;
    int d = threadIdx.x;                       // 64 threads
    const float* ne = node_emb + (long)local_nodes[i] * DN_;
    float acc = 0.0f;
    for (int k = 0; k < DN_; ++k) acc += ne[k] * gat_w[k * DL_ + d];
    h[i * DL_ + d] = acc;
    __shared__ float s1[DL_], s2[DL_];
    s1[d] = acc * a1[d]; s2[d] = acc * a2[d];
    __syncthreads();
    if (d == 0) {
        float x = 0, y = 0;
        for (int j = 0; j < DL_; ++j) { x += s1[j]; y += s2[j]; }
        e1[i] = x; e2[i] = y;
    }
}

// ---------------- GAT attention row ----------------
__global__ void k_gat_attn(const float* adj, const float* e1, const float* e2,
                           const float* h, float* gat)
{
    int i = blockIdx.x;
    int tid = threadIdx.x;                     // 256 threads
    __shared__ float w[NLL];
    __shared__ float red[256];
    float e1i = e1[i];
    float mx = -3.0e38f;
    for (int j = tid; j < NLL; j += 256) {
        float a = adj[(long)i * NLL + j];
        float x = e1i + e2[j]; x = x > 0.0f ? x : 0.2f * x;
        float e = (a > 0.0f) ? x : -1.0e30f;
        w[j] = e; mx = fmaxf(mx, e);
    }
    red[tid] = mx; __syncthreads();
    for (int s = 128; s > 0; s >>= 1) { if (tid < s) red[tid] = fmaxf(red[tid], red[tid + s]); __syncthreads(); }
    mx = red[0]; __syncthreads();
    float sm = 0.0f;
    for (int j = tid; j < NLL; j += 256) { float ex = __expf(w[j] - mx); w[j] = ex; sm += ex; }
    red[tid] = sm; __syncthreads();
    for (int s = 128; s > 0; s >>= 1) { if (tid < s) red[tid] += red[tid + s]; __syncthreads(); }
    sm = red[0]; __syncthreads();
    if (tid < DL_) {
        float acc = 0.0f;
        for (int j = 0; j < NLL; ++j) acc += w[j] * h[j * DL_ + tid];
        acc /= sm;
        gat[i * DL_ + tid] = acc > 0.0f ? acc : (__expf(acc) - 1.0f);
    }
}

// ---------------- mean of valid node embeddings per batch ----------------
__global__ void k_mean(const int* node_traj, const int* seq_len, const float* node_emb, float* smean)
{
    int b = blockIdx.x, d = threadIdx.x;       // 128 threads
    int lens = max(seq_len[b], 1);
    float acc = 0.0f;
    for (int l = 0; l < lens; ++l)
        acc += node_emb[(long)node_traj[b * LL + l] * DN_ + d];
    smean[b * DN_ + d] = acc / (float)lens;
}

// ---------------- key padding bias ----------------
__global__ void k_keybias(const int* seq_len, float* kb)
{
    int i = blockIdx.x * 256 + threadIdx.x;
    if (i >= BB * LL) return;
    int b = i / LL, l = i % LL;
    int lens = max(seq_len[b], 1);
    kb[i] = (l < lens) ? 0.0f : -1.0e30f;
}

// ---------------- build embeddings + PE -> src ----------------
__global__ void k_build_emb(const int* node_traj, const int* r_traj, const int* seq_len,
                            const int* local_idx, const float* node_emb, const float* dir_emb,
                            const float* gat, const float* nse, const float* smean,
                            const float* fuse_w, const float* fuse_b,
                            float* src_f, bf16* src_bf)
{
    int t = blockIdx.x;
    int b = t / LL, l = t % LL;
    int tid = threadIdx.x;                     // 128 threads
    __shared__ float swh[DG_], ss[DN_], sfu[DG_];
    int node = node_traj[t];
    int lens = max(seq_len[b], 1);
    bool valid = l < lens;
    if (tid < DG_) swh[tid] = nse[(long)node * DG_ + tid];
    if (tid < DN_) ss[tid] = smean[b * DN_ + tid];
    __syncthreads();
    if (tid < DG_) {
        const float* fw = fuse_w + tid * (DG_ + DN_);
        float acc = fuse_b[tid];
        for (int j = 0; j < DG_; ++j) acc += swh[j] * fw[j];
        for (int j = 0; j < DN_; ++j) acc += ss[j] * fw[DG_ + j];
        sfu[tid] = acc > 0.0f ? acc : 0.2f * acc;
    }
    __syncthreads();
    int r = r_traj[t], li = local_idx[t];
    for (int d = tid; d < MM; d += blockDim.x) {
        float v;
        if (d < DN_)             v = node_emb[(long)node * DN_ + d];
        else if (d < DN_ + DD_)  v = dir_emb[r * DD_ + (d - DN_)];
        else if (d < DN_ + DD_ + DL_) v = gat[li * DL_ + (d - DN_ - DD_)];
        else                     v = sfu[d - (DN_ + DD_ + DL_)];
        if (!valid) v = -1.0f;
        float arg = (float)b * __expf(-(float)(d & ~1) * (logf(10000.0f) / (float)MM));
        float pe = (d & 1) ? __cosf(arg) : __sinf(arg);
        float s = v + pe;
        src_f[(long)t * MM + d] = s;
        src_bf[(long)t * MM + d] = (bf16)s;
    }
}

// ---------------- decoder target token = src row at last valid pos ----------------
__global__ void k_tgt(const float* src_f, const int* seq_len, float* t_f, bf16* t_bf)
{
    int b = blockIdx.x, d = threadIdx.x;       // MM threads
    int lens = max(seq_len[b], 1);
    float v = src_f[((long)b * LL + lens - 1) * MM + d];
    t_f[b * MM + d] = v; t_bf[b * MM + d] = (bf16)v;
}

// ---------------- LayerNorm: y = LN(x [+ res]) * g + b ; one wave per row ----------------
__global__ void k_ln(const float* x, const float* res, const float* g, const float* bta,
                     float* oF, bf16* oB, int rows, int n)
{
    int wave = threadIdx.x >> 5, lane = threadIdx.x & 31;
    int row = blockIdx.x * (blockDim.x >> 5) + wave;
    if (row >= rows) return;
    const float* xp = x + (long)row * n;
    const float* rp = res ? res + (long)row * n : nullptr;
    float vals[10];
    int cnt = 0; float s = 0.0f;
    for (int i = lane; i < n; i += 32) { float v = xp[i] + (rp ? rp[i] : 0.0f); vals[cnt++] = v; s += v; }
    s = wred_sum(s);
    float mean = s / (float)n;
    float vs = 0.0f;
    for (int c = 0; c < cnt; ++c) { float d = vals[c] - mean; vs += d * d; }
    vs = wred_sum(vs) / (float)n;
    float inv = rsqrtf(vs + 1e-5f);
    int c = 0;
    for (int i = lane; i < n; i += 32) {
        float y = (vals[c++] - mean) * inv * g[i] + bta[i];
        if (oF) oF[(long)row * n + i] = y;
        if (oB) oB[(long)row * n + i] = (bf16)y;
    }
}

// ---------------- row softmax (fp32 scores -> bf16 probs) ----------------
__global__ void k_softmax(const float* sc, bf16* pr, long rows, int n)
{
    int wave = threadIdx.x >> 5, lane = threadIdx.x & 31;
    long row = (long)blockIdx.x * (blockDim.x >> 5) + wave;
    if (row >= rows) return;
    const float* sp = sc + row * n;
    float v[16]; int cnt = 0; float mx = -3.0e38f;
    for (int i = lane; i < n; i += 32) { float x = sp[i]; v[cnt++] = x; mx = fmaxf(mx, x); }
    mx = wred_max(mx);
    float sm = 0.0f;
    for (int c = 0; c < cnt; ++c) { v[c] = __expf(v[c] - mx); sm += v[c]; }
    sm = wred_sum(sm);
    float inv = 1.0f / sm;
    int c = 0;
    for (int i = lane; i < n; i += 32) pr[row * n + i] = (bf16)(v[c++] * inv);
}

// ---------------- decoder cross-attention (64 queries) ----------------
__global__ void k_xattn(const float* q, const float* Kf, const float* Vf, const float* kb,
                        float* oF, bf16* oB)
{
    int z = blockIdx.x; int b = z / NHH, h2 = z % NHH;
    __shared__ float qv[DHH], sc[LL], red[256];
    int tid = threadIdx.x;                     // 256 threads
    if (tid < DHH) qv[tid] = q[b * MM + h2 * DHH + tid];
    __syncthreads();
    float scale = rsqrtf((float)DHH);
    for (int k = tid; k < LL; k += 256) {
        const float* kp = Kf + ((long)b * LL + k) * MM + h2 * DHH;
        float acc = 0.0f;
        for (int d = 0; d < DHH; ++d) acc += qv[d] * kp[d];
        sc[k] = acc * scale + kb[b * LL + k];
    }
    __syncthreads();
    float mx = -3.0e38f;
    for (int k = tid; k < LL; k += 256) mx = fmaxf(mx, sc[k]);
    red[tid] = mx; __syncthreads();
    for (int s = 128; s > 0; s >>= 1) { if (tid < s) red[tid] = fmaxf(red[tid], red[tid + s]); __syncthreads(); }
    mx = red[0]; __syncthreads();
    float sm = 0.0f;
    for (int k = tid; k < LL; k += 256) { float e = __expf(sc[k] - mx); sc[k] = e; sm += e; }
    red[tid] = sm; __syncthreads();
    for (int s = 128; s > 0; s >>= 1) { if (tid < s) red[tid] += red[tid + s]; __syncthreads(); }
    sm = red[0]; __syncthreads();
    if (tid < DHH) {
        float acc = 0.0f;
        for (int k = 0; k < LL; ++k) acc += sc[k] * Vf[((long)b * LL + k) * MM + h2 * DHH + tid];
        float o = acc / sm;
        oF[b * MM + h2 * DHH + tid] = o;
        oB[b * MM + h2 * DHH + tid] = (bf16)o;
    }
}

// ---------------- concat decoder output with taxi embedding ----------------
__global__ void k_concat(const float* t, const int* taxi_id, const float* taxi_emb, bf16* cat)
{
    int b = blockIdx.x, d = threadIdx.x;       // KCAT threads
    float v = (d < MM) ? t[b * MM + d] : taxi_emb[taxi_id[b] * DT_ + (d - MM)];
    cat[b * KCAT + d] = (bf16)v;
}

// ---------------- final epilogue: logits += trans_map[last_node] ----------------
__global__ void k_transadd(float* out, const float* trans, const int* node_traj, const int* seq_len)
{
    long i = (long)blockIdx.x * 256 + threadIdx.x;
    if (i >= (long)BB * NNODE) return;
    int b = (int)(i / NNODE), n = (int)(i % NNODE);
    int lens = max(seq_len[b], 1);
    int last = node_traj[b * LL + lens - 1];
    out[i] += trans[(long)last * NNODE + n];
}

// =====================================================================
extern "C" void kernel_launch(void* const* d_in, const int* in_sizes, int n_in,
                              void* d_out, int out_size, void* d_ws, size_t ws_size,
                              hipStream_t stream)
{
    (void)in_sizes; (void)n_in; (void)out_size; (void)ws_size;
    // ---- inputs ----
    const int*   node_traj = (const int*)d_in[0];
    const int*   r_traj    = (const int*)d_in[1];
    const int*   taxi_id   = (const int*)d_in[2];
    const int*   seq_len   = (const int*)d_in[3];
    const int*   local_idx = (const int*)d_in[4];
    const int*   local_nodes = (const int*)d_in[5];
    const float* adj       = (const float*)d_in[6];
    const float* nse       = (const float*)d_in[7];
    const float* trans     = (const float*)d_in[8];
    // ---- params (dict insertion order) ----
    const float* node_emb  = (const float*)d_in[9];
    const float* dir_emb   = (const float*)d_in[10];
    const float* taxi_emb  = (const float*)d_in[11];
    const float* fuse_w    = (const float*)d_in[12];
    const float* fuse_b    = (const float*)d_in[13];
    const float* gat_w     = (const float*)d_in[14];
    const float* gat_a1    = (const float*)d_in[15];
    const float* gat_a2    = (const float*)d_in[16];
    const float* enc_qkv_w = (const float*)d_in[17];
    const float* enc_qkv_b = (const float*)d_in[18];
    const float* enc_out_w = (const float*)d_in[19];
    const float* enc_out_b = (const float*)d_in[20];
    const float* enc_ln1_g = (const float*)d_in[21];
    const float* enc_ln1_b = (const float*)d_in[22];
    const float* enc_ln2_g = (const float*)d_in[23];
    const float* enc_ln2_b = (const float*)d_in[24];
    const float* enc_ff1_w = (const float*)d_in[25];
    const float* enc_ff1_b = (const float*)d_in[26];
    const float* enc_ff2_w = (const float*)d_in[27];
    const float* enc_ff2_b = (const float*)d_in[28];
    const float* enc_ng    = (const float*)d_in[29];
    const float* enc_nb    = (const float*)d_in[30];
    const float* sa_qkv_w  = (const float*)d_in[31];
    const float* sa_qkv_b  = (const float*)d_in[32];
    const float* sa_out_w  = (const float*)d_in[33];
    const float* sa_out_b  = (const float*)d_in[34];
    const float* ca_qkv_w  = (const float*)d_in[35];
    const float* ca_qkv_b  = (const float*)d_in[36];
    const float* ca_out_w  = (const float*)d_in[37];
    const float* ca_out_b  = (const float*)d_in[38];
    const float* dec_ln1_g = (const float*)d_in[39];
    const float* dec_ln1_b = (const float*)d_in[40];
    const float* dec_ln2_g = (const float*)d_in[41];
    const float* dec_ln2_b = (const float*)d_in[42];
    const float* dec_ln3_g = (const float*)d_in[43];
    const float* dec_ln3_b = (const float*)d_in[44];
    const float* dec_ff1_w = (const float*)d_in[45];
    const float* dec_ff1_b = (const float*)d_in[46];
    const float* dec_ff2_w = (const float*)d_in[47];
    const float* dec_ff2_b = (const float*)d_in[48];
    const float* dec_ng    = (const float*)d_in[49];
    const float* dec_nb    = (const float*)d_in[50];
    const float* fc_w      = (const float*)d_in[51];
    const float* fc_b      = (const float*)d_in[52];

    // ---- workspace bump allocator ----
    char* wp = (char*)d_ws;
    auto alloc = [&](size_t elems, size_t esz) -> void* {
        void* r = (void*)wp;
        wp += ((elems * esz) + 255) & ~(size_t)255;
        return r;
    };
    auto af = [&](size_t n) { return (float*)alloc(n, 4); };
    auto ab = [&](size_t n) { return (bf16*)alloc(n, 2); };

    const long TOK = (long)BB * LL;
    const long ZH  = (long)BB * NHH;

    // bf16 weight copies
    bf16* qkvw_bf = ab(2 * 960 * MM);
    bf16* outw_bf = ab(2 * MM * MM);
    bf16* ff1w_bf = ab(2 * (size_t)FFF * MM);
    bf16* ff2w_bf = ab(2 * (size_t)MM * FFF);
    bf16* saq_bf  = ab(960 * MM);
    bf16* sao_bf  = ab(MM * MM);
    bf16* caq_bf  = ab(960 * MM);
    bf16* cao_bf  = ab(MM * MM);
    bf16* dff1_bf = ab((size_t)FFF * MM);
    bf16* dff2_bf = ab((size_t)MM * FFF);
    bf16* fcw_bf  = ab((size_t)(NNODE + 16) * KCAT);   // +16 rows slack for 32-wide N tiles

    // activations
    float* h_f     = af(NLL * DL_);
    float* e1_f    = af(NLL);
    float* e2_f    = af(NLL);
    float* gat_f   = af(NLL * DL_);
    float* smean   = af(BB * DN_);
    float* kb      = af(BB * LL);
    float* src_f   = af(TOK * MM);
    bf16*  src_bf  = ab(TOK * MM);
    float* x_f     = af(TOK * MM);
    bf16*  x_bf    = ab(TOK * MM);
    float* tmp_f   = af(TOK * MM);
    bf16*  qkv_bf  = ab(TOK * 960);
    bf16*  qp      = ab(ZH * LL * DHP);
    bf16*  kpad    = ab(ZH * LL * DHP);
    bf16*  vpT     = ab(ZH * DHP * LL);
    float* scores_f = af(ZH * LL * LL);
    bf16*  probs_bf = ab(ZH * LL * LL);
    bf16*  attn_bf  = ab(TOK * MM);
    bf16*  ff_bf    = ab(TOK * (size_t)FFF);
    bf16*  mem_bf   = ab(TOK * MM);
    float* Kf       = af(TOK * MM);
    float* Vf       = af(TOK * MM);
    float* t_f      = af(BB * MM);
    bf16*  t_bf     = ab(BB * MM);
    bf16*  d1_bf    = ab(BB * MM);
    float* d2_f     = af(BB * MM);
    float* q_f      = af(BB * MM);
    float* ca_f     = af(BB * MM);
    bf16*  ca_bf    = ab(BB * MM);
    bf16*  dff_bf   = ab(BB * FFF);
    bf16*  cat_bf   = ab(BB * KCAT);

    auto cvt = [&](const float* s, bf16* dst, long n) {
        k_cvt<<<dim3((unsigned)((n + 255) / 256)), 256, 0, stream>>>(s, dst, n);
    };
    auto gtn = [&](const bf16* A, long lda, long a1, long a2,
                   const bf16* W, long ldw, long w1, long w2,
                   const float* bias, const float* colb, long cb1, long cb2,
                   float alpha, int relu, float* oF, bf16* oB, long ldc, long cs1, long cs2,
                   int Mr, int Nc, int K, int Z, int zdiv) {
        dim3 g((Nc + 31) / 32, (Mr + 127) / 128, Z);
        gemm_tn<<<g, 128, 0, stream>>>(A, lda, a1, a2, W, ldw, w1, w2,
                                       bias, colb, cb1, cb2, alpha, relu,
                                       oF, oB, ldc, cs1, cs2, Mr, Nc, K, zdiv);
    };
    auto ln = [&](const float* x, const float* res, const float* g, const float* b,
                  float* oF, bf16* oB, int rows) {
        k_ln<<<dim3((rows + 7) / 8), 256, 0, stream>>>(x, res, g, b, oF, oB, rows, MM);
    };

    // ---- convert weights once ----
    cvt(enc_qkv_w, qkvw_bf, 2L * 960 * MM);
    cvt(enc_out_w, outw_bf, 2L * MM * MM);
    cvt(enc_ff1_w, ff1w_bf, 2L * FFF * MM);
    cvt(enc_ff2_w, ff2w_bf, 2L * MM * FFF);
    cvt(sa_qkv_w, saq_bf, 960L * MM);
    cvt(sa_out_w, sao_bf, (long)MM * MM);
    cvt(ca_qkv_w, caq_bf, 960L * MM);
    cvt(ca_out_w, cao_bf, (long)MM * MM);
    cvt(dec_ff1_w, dff1_bf, (long)FFF * MM);
    cvt(dec_ff2_w, dff2_bf, (long)MM * FFF);
    cvt(fc_w, fcw_bf, (long)NNODE * KCAT);

    // ---- embedding stage ----
    k_gat_prep<<<dim3(NLL), DL_, 0, stream>>>(local_nodes, node_emb, gat_w, gat_a1, gat_a2, h_f, e1_f, e2_f);
    k_gat_attn<<<dim3(NLL), 256, 0, stream>>>(adj, e1_f, e2_f, h_f, gat_f);
    k_mean<<<dim3(BB), DN_, 0, stream>>>(node_traj, seq_len, node_emb, smean);
    k_keybias<<<dim3((BB * LL + 255) / 256), 256, 0, stream>>>(seq_len, kb);
    k_build_emb<<<dim3((unsigned)TOK), 128, 0, stream>>>(node_traj, r_traj, seq_len, local_idx,
                                                         node_emb, dir_emb, gat_f, nse, smean,
                                                         fuse_w, fuse_b, src_f, src_bf);
    k_tgt<<<dim3(BB), MM, 0, stream>>>(src_f, seq_len, t_f, t_bf);

    // ---- encoder (2 layers) ----
    const float* cur_f = src_f;
    const bf16*  cur_bf = src_bf;
    const float inv_sqrt_dh = 1.0f / sqrtf((float)DHH);
    const long splitN = ZH * LL * DHP;
    for (int layer = 0; layer < 2; ++layer) {
        // QKV
        gtn(cur_bf, MM, 0, 0, qkvw_bf + (size_t)layer * 960 * MM, MM, 0, 0,
            enc_qkv_b + layer * 960, nullptr, 0, 0, 1.0f, 0,
            nullptr, qkv_bf, 960, 0, 0, (int)TOK, 960, MM, 1, 1);
        // split into padded heads (q, k row-major; v transposed)
        k_split_qkv<<<dim3((unsigned)((splitN + 255) / 256)), 256, 0, stream>>>(qkv_bf, qp, kpad, vpT);
        // scores = Q·K^T/sqrt(dh) + key_bias   (z = b*NH + h)
        gtn(qp, DHP, (long)NHH * LL * DHP, (long)LL * DHP,
            kpad, DHP, (long)NHH * LL * DHP, (long)LL * DHP,
            nullptr, kb, LL, 0, inv_sqrt_dh, 0,
            scores_f, nullptr, LL, (long)NHH * LL * LL, (long)LL * LL,
            LL, LL, DHP, (int)ZH, NHH);
        // softmax rows
        k_softmax<<<dim3((BB * NHH * LL) / 8), 256, 0, stream>>>(scores_f, probs_bf, (long)BB * NHH * LL, LL);
        // O = P·V via TN gemm against transposed V; heads merged into [B*L, M]
        gtn(probs_bf, LL, (long)NHH * LL * LL, (long)LL * LL,
            vpT, LL, (long)NHH * DHP * LL, (long)DHP * LL,
            nullptr, nullptr, 0, 0, 1.0f, 0,
            nullptr, attn_bf, MM, (long)LL * MM, DHH,
            LL, DHH, LL, (int)ZH, NHH);
        // out projection
        gtn(attn_bf, MM, 0, 0, outw_bf + (size_t)layer * MM * MM, MM, 0, 0,
            enc_out_b + layer * MM, nullptr, 0, 0, 1.0f, 0,
            tmp_f, nullptr, MM, 0, 0, (int)TOK, MM, MM, 1, 1);
        ln(cur_f, tmp_f, enc_ln1_g + layer * MM, enc_ln1_b + layer * MM, x_f, x_bf, (int)TOK);
        // FFN
        gtn(x_bf, MM, 0, 0, ff1w_bf + (size_t)layer * FFF * MM, MM, 0, 0,
            enc_ff1_b + layer * FFF, nullptr, 0, 0, 1.0f, 1,
            nullptr, ff_bf, FFF, 0, 0, (int)TOK, FFF, MM, 1, 1);
        gtn(ff_bf, FFF, 0, 0, ff2w_bf + (size_t)layer * MM * FFF, FFF, 0, 0,
            enc_ff2_b + layer * MM, nullptr, 0, 0, 1.0f, 0,
            tmp_f, nullptr, MM, 0, 0, (int)TOK, MM, FFF, 1, 1);
        ln(x_f, tmp_f, enc_ln2_g + layer * MM, enc_ln2_b + layer * MM, x_f, x_bf, (int)TOK);
        cur_f = x_f; cur_bf = x_bf;
    }
    ln(cur_f, nullptr, enc_ng, enc_nb, nullptr, mem_bf, (int)TOK);

    // ---- decoder ----
    // self-attn over single key reduces to o = v
    gtn(t_bf, MM, 0, 0, saq_bf + 2 * MM * MM, MM, 0, 0,
        sa_qkv_b + 2 * MM, nullptr, 0, 0, 1.0f, 0,
        nullptr, d1_bf, MM, 0, 0, BB, MM, MM, 1, 1);
    gtn(d1_bf, MM, 0, 0, sao_bf, MM, 0, 0, sa_out_b, nullptr, 0, 0, 1.0f, 0,
        d2_f, nullptr, MM, 0, 0, BB, MM, MM, 1, 1);
    ln(t_f, d2_f, dec_ln1_g, dec_ln1_b, t_f, t_bf, BB);
    // cross-attn
    gtn(t_bf, MM, 0, 0, caq_bf, MM, 0, 0, ca_qkv_b, nullptr, 0, 0, 1.0f, 0,
        q_f, nullptr, MM, 0, 0, BB, MM, MM, 1, 1);
    gtn(mem_bf, MM, 0, 0, caq_bf + (size_t)MM * MM, MM, 0, 0,
        ca_qkv_b + MM, nullptr, 0, 0, 1.0f, 0,
        Kf, nullptr, MM, 0, 0, (int)TOK, MM, MM, 1, 1);
    gtn(mem_bf, MM, 0, 0, caq_bf + 2 * (size_t)MM * MM, MM, 0, 0,
        ca_qkv_b + 2 * MM, nullptr, 0, 0, 1.0f, 0,
        Vf, nullptr, MM, 0, 0, (int)TOK, MM, MM, 1, 1);
    k_xattn<<<dim3(BB * NHH), 256, 0, stream>>>(q_f, Kf, Vf, kb, ca_f, ca_bf);
    gtn(ca_bf, MM, 0, 0, cao_bf, MM, 0, 0, ca_out_b, nullptr, 0, 0, 1.0f, 0,
        d2_f, nullptr, MM, 0, 0, BB, MM, MM, 1, 1);
    ln(t_f, d2_f, dec_ln2_g, dec_ln2_b, t_f, t_bf, BB);
    // decoder FFN
    gtn(t_bf, MM, 0, 0, dff1_bf, MM, 0, 0, dec_ff1_b, nullptr, 0, 0, 1.0f, 1,
        nullptr, dff_bf, FFF, 0, 0, BB, FFF, MM, 1, 1);
    gtn(dff_bf, FFF, 0, 0, dff2_bf, FFF, 0, 0, dec_ff2_b, nullptr, 0, 0, 1.0f, 0,
        d2_f, nullptr, MM, 0, 0, BB, MM, FFF, 1, 1);
    ln(t_f, d2_f, dec_ln3_g, dec_ln3_b, t_f, t_bf, BB);
    ln(t_f, nullptr, dec_ng, dec_nb, t_f, t_bf, BB);

    // ---- head ----
    k_concat<<<dim3(BB), KCAT, 0, stream>>>(t_f, taxi_id, taxi_emb, cat_bf);
    gtn(cat_bf, KCAT, 0, 0, fcw_bf, KCAT, 0, 0, fc_b, nullptr, 0, 0, 1.0f, 0,
        (float*)d_out, nullptr, NNODE, 0, 0, BB, NNODE, KCAT, 1, 1);
    k_transadd<<<dim3((BB * NNODE + 255) / 256), 256, 0, stream>>>((float*)d_out, trans, node_traj, seq_len);
}
